// EpigeneticNormalization_60447369724344
// MI455X (gfx1250) — matrix operand order repaired and checked by reading, not compiled, not run on hardware
//
#include <hip/hip_runtime.h>
#include <hip/hip_bf16.h>

// ---------------------------------------------------------------------------
// MI455X (gfx1250) implementation, round 2.
// bf16 V_WMMA_F32_16X16X32_BF16 for all three GEMMs (compute-bound: ~240 GFLOP
// vs ~0.6 GB HBM), fp32 accumulate, fused epilogues.
// Changes vs round 1:
//  * trans_mlp1/2 use a 32-token M tile: each B fragment feeds 2 WMMAs
//    (16 wmma : 20 ds_load_b128 per wave per K-step, was 8:18), and weight
//    re-reads from L2 are halved.
//  * silu/softmax use v_rcp_f32 (__builtin_amdgcn_rcpf) instead of precise
//    division (kills the v_div_scale/v_div_fmas chains seen in round 1).
//  * post-K-loop LDS buffers (pred hbuf, mlp2 LN tile) alias the A/B staging
//    area (dead after the last barrier) -> smaller LDS footprint, better
//    WGP occupancy.
// Workspace (~199 MB): sid | w1b | tw1b | tw2b | t(bf16) | y1(bf16)
// ---------------------------------------------------------------------------

typedef unsigned short u16;
typedef __attribute__((ext_vector_type(16))) __bf16 v16bf;
typedef __attribute__((ext_vector_type(8)))  float  v8f;
typedef __attribute__((ext_vector_type(4)))  unsigned short us4;

#define NTOK 32768
#define DIM  1024
#define HID  512
#define TDIM 2048
#define A_STRIDE  40   // bf16 elems; 8-elem runs stay 16B aligned
#define BT_STRIDE 40
#define YT_STRIDE 1028 // fp32 elems; 1028%64==4 -> conflict-free LN tile

__device__ __forceinline__ u16 f2bf(float f) {
  union { float f; unsigned u; } v; v.f = f;
  unsigned r = v.u + 0x7FFFu + ((v.u >> 16) & 1u);  // round-to-nearest-even
  return (u16)(r >> 16);
}

__device__ __forceinline__ float fast_rcp(float x) {
  return __builtin_amdgcn_rcpf(x);                  // v_rcp_f32
}

__device__ __forceinline__ float silu(float v) {
  return v * fast_rcp(1.0f + __expf(-v));
}

union Frag16 { v16bf v; u16 u[16]; };

// A fragment (16x32 bf16, MxK): lane L -> row M=L&15, K-base 8*(L>=16);
// elems 0..7 = K kb..kb+7, elems 8..15 = K kb+16..kb+23  (ISA 7.12.2)
__device__ __forceinline__ v16bf load_a_frag(const u16* As, int lane) {
  int m  = lane & 15;
  int kb = (lane >> 4) << 3;
  const u16* p = As + m * A_STRIDE + kb;
  Frag16 f;
#pragma unroll
  for (int i = 0; i < 8; ++i) { f.u[i] = p[i]; f.u[8 + i] = p[16 + i]; }
  return f.v;
}

// B fragment (32x16 bf16, KxN), Bt staged transposed in LDS as [N][K]:
// lane L -> col N=n0+(L&15), K-base 8*(L>=16); elems 0..7 = K kb..kb+7, elems 8..15 = K kb+16..kb+23
__device__ __forceinline__ v16bf load_b_frag(const u16* Bt, int lane, int n0) {
  int n  = n0 + (lane & 15);
  int kh = (lane >> 4) << 3;
  const u16* p = Bt + n * BT_STRIDE + kh;
  Frag16 f;
#pragma unroll
  for (int i = 0; i < 8; ++i) { f.u[i] = p[i]; f.u[8 + i] = p[16 + i]; }
  return f.v;
}

__device__ __forceinline__ v8f wmma_bf16(v16bf a, v16bf b, v8f c) {
  // (neg_a, A, neg_b, B, c_mod, C, reuse_a, reuse_b)
  return __builtin_amdgcn_wmma_f32_16x16x32_bf16(false, a, false, b, (short)0, c,
                                                 false, false);
}

// ---------------------------------------------------------------------------
__global__ void cvt_bf16_kernel(const float* __restrict__ s, u16* __restrict__ d,
                                int n) {
  int i = blockIdx.x * blockDim.x + threadIdx.x;
  int stride = gridDim.x * blockDim.x;
  for (; i < n; i += stride) d[i] = f2bf(s[i]);
}

// ---------------------------------------------------------------------------
// pred: h = silu(x @ pred_w1 + b1); logits = h @ pred_w2 + b2; softmax; argmax
// Block = 256 thr (8 waves), 16 tokens/block, each wave owns 64 cols of h.
// hbuf aliases the A/B staging area (dead after the K loop's last barrier).
__global__ __launch_bounds__(256) void pred_kernel(
    const float* __restrict__ x, const u16* __restrict__ w1b,
    const float* __restrict__ b1, const float* __restrict__ w2,
    const float* __restrict__ b2, float* __restrict__ states,
    int* __restrict__ sid) {
  extern __shared__ unsigned char smem[];
  u16*   As   = (u16*)smem;                                 // 16*40 bf16
  u16*   Bt   = (u16*)(smem + 16 * A_STRIDE * 2);           // 512*40 bf16
  float* hbuf = (float*)smem;                               // 16*520 f32 (alias)

  const int tid  = threadIdx.x;
  const int lane = tid & 31;
  const int wave = tid >> 5;
  const int tok0 = blockIdx.x * 16;

  v8f acc[4] = {};

  for (int k0 = 0; k0 < DIM; k0 += 32) {
    for (int e = tid; e < 16 * 32; e += 256) {
      int m = e >> 5, k = e & 31;
      As[m * A_STRIDE + k] = f2bf(x[(size_t)(tok0 + m) * DIM + k0 + k]);
    }
    for (int q = tid; q < 32 * (HID / 4); q += 256) {   // 4096 8B quads
      int k = q >> 7, n4 = (q & 127) << 2;
      us4 w = *(const us4*)(w1b + (size_t)(k0 + k) * HID + n4);
      Bt[(n4 + 0) * BT_STRIDE + k] = w.x;
      Bt[(n4 + 1) * BT_STRIDE + k] = w.y;
      Bt[(n4 + 2) * BT_STRIDE + k] = w.z;
      Bt[(n4 + 3) * BT_STRIDE + k] = w.w;
    }
    __syncthreads();
    v16bf a = load_a_frag(As, lane);
#pragma unroll
    for (int j = 0; j < 4; ++j) {
      v16bf b = load_b_frag(Bt, lane, wave * 64 + j * 16);
      acc[j] = wmma_bf16(a, b, acc[j]);
    }
    __syncthreads();          // As/Bt dead after this -> hbuf may alias
  }

  // epilogue: bias + silu -> hbuf  (D elem i -> row (L>=16)*8+i, col L&15)
#pragma unroll
  for (int j = 0; j < 4; ++j) {
    int n = wave * 64 + j * 16 + (lane & 15);
    float bias = b1[n];
    int mb = (lane >> 4) << 3;
#pragma unroll
    for (int i = 0; i < 8; ++i)
      hbuf[(mb + i) * 520 + n] = silu(acc[j][i] + bias);
  }
  __syncthreads();

  // logits (512 -> 3): 16 lanes per token, strided partials + xor reduction
  int tok = tid >> 4;
  int jj  = tid & 15;
  float l0 = 0.f, l1 = 0.f, l2 = 0.f;
  for (int f = jj; f < HID; f += 16) {
    float hv = hbuf[tok * 520 + f];
    l0 += hv * w2[f * 3 + 0];
    l1 += hv * w2[f * 3 + 1];
    l2 += hv * w2[f * 3 + 2];
  }
#pragma unroll
  for (int off = 8; off; off >>= 1) {
    l0 += __shfl_xor(l0, off);
    l1 += __shfl_xor(l1, off);
    l2 += __shfl_xor(l2, off);
  }
  if (jj == 0) {
    l0 += b2[0]; l1 += b2[1]; l2 += b2[2];
    float mx = fmaxf(l0, fmaxf(l1, l2));
    float e0 = __expf(l0 - mx), e1 = __expf(l1 - mx), e2 = __expf(l2 - mx);
    float inv = fast_rcp(e0 + e1 + e2);
    size_t tg = (size_t)tok0 + tok;
    states[tg * 3 + 0] = e0 * inv;
    states[tg * 3 + 1] = e1 * inv;
    states[tg * 3 + 2] = e2 * inv;
    int id = 0; float best = l0;
    if (l1 > best) { best = l1; id = 1; }
    if (l2 > best) { id = 2; }
    sid[tg] = id;
  }
}

// ---------------------------------------------------------------------------
// ln_gather: xn = LN(x)*gamma[sid]+beta[sid] -> d_out ; t = bf16[xn, emb[sid]]
// One wave per token (32 lanes x 32 elems).
__global__ __launch_bounds__(256) void ln_gather_kernel(
    const float* __restrict__ x, const int* __restrict__ sid,
    const float* __restrict__ g, const float* __restrict__ b,
    const float* __restrict__ emb, float* __restrict__ out,
    u16* __restrict__ t) {
  const int wave = threadIdx.x >> 5, lane = threadIdx.x & 31;
  const size_t tok = (size_t)blockIdx.x * 8 + wave;
  const float* xr = x + tok * DIM;
  float vals[32];
  float s = 0.f, sq = 0.f;
#pragma unroll
  for (int i = 0; i < 32; ++i) {
    float v = xr[lane + i * 32];
    vals[i] = v; s += v; sq += v * v;
  }
#pragma unroll
  for (int off = 16; off; off >>= 1) {
    s  += __shfl_xor(s, off);
    sq += __shfl_xor(sq, off);
  }
  float m    = s * (1.0f / DIM);
  float var  = sq * (1.0f / DIM) - m * m;
  float rstd = rsqrtf(var + 1e-5f);
  int id = sid[tok];
  const float* gr = g   + (size_t)id * DIM;
  const float* br = b   + (size_t)id * DIM;
  const float* er = emb + (size_t)id * DIM;
  float* orow = out + tok * DIM;
  u16*   trow = t   + tok * TDIM;
#pragma unroll
  for (int i = 0; i < 32; ++i) {
    int c = lane + i * 32;
    float xn = (vals[i] - m) * rstd * gr[c] + br[c];
    orow[c]       = xn;              // keep xn for final residual add
    trow[c]       = f2bf(xn);
    trow[DIM + c] = f2bf(er[c]);
  }
}

// ---------------------------------------------------------------------------
// trans_mlp1: y1 = silu(t @ tr_w1 + b1), [N,2048]x[2048,1024], bf16 out.
// M tile = 32 tokens: each B fragment is reused by 2 WMMAs.
__global__ __launch_bounds__(256) void trans_mlp1_kernel(
    const u16* __restrict__ t, const u16* __restrict__ w,
    const float* __restrict__ bias, u16* __restrict__ y1) {
  extern __shared__ unsigned char smem[];
  u16* As = (u16*)smem;                         // 32*40 bf16
  u16* Bt = (u16*)(smem + 32 * A_STRIDE * 2);   // 1024*40 bf16
  const int tid = threadIdx.x, lane = tid & 31, wave = tid >> 5;
  const int tok0 = blockIdx.x * 32;
  v8f acc[2][8] = {};
  for (int k0 = 0; k0 < TDIM; k0 += 32) {
    for (int e = tid; e < 32 * 32; e += 256) {
      int m = e >> 5, k = e & 31;
      As[m * A_STRIDE + k] = t[(size_t)(tok0 + m) * TDIM + k0 + k];
    }
    for (int q = tid; q < 32 * (DIM / 4); q += 256) {   // 8192 8B quads
      int k = q >> 8, n4 = (q & 255) << 2;
      us4 wv = *(const us4*)(w + (size_t)(k0 + k) * DIM + n4);
      Bt[(n4 + 0) * BT_STRIDE + k] = wv.x;
      Bt[(n4 + 1) * BT_STRIDE + k] = wv.y;
      Bt[(n4 + 2) * BT_STRIDE + k] = wv.z;
      Bt[(n4 + 3) * BT_STRIDE + k] = wv.w;
    }
    if (k0 + 32 < TDIM)   // global_prefetch_b8 next weight slab
      __builtin_prefetch(w + (size_t)(k0 + 32) * DIM + ((size_t)tid << 2), 0, 1);
    __syncthreads();
    v16bf a0 = load_a_frag(As, lane);
    v16bf a1 = load_a_frag(As + 16 * A_STRIDE, lane);
#pragma unroll
    for (int j = 0; j < 8; ++j) {
      v16bf b = load_b_frag(Bt, lane, wave * 128 + j * 16);
      acc[0][j] = wmma_bf16(a0, b, acc[0][j]);
      acc[1][j] = wmma_bf16(a1, b, acc[1][j]);
    }
    __syncthreads();
  }
#pragma unroll
  for (int mt = 0; mt < 2; ++mt) {
#pragma unroll
    for (int j = 0; j < 8; ++j) {
      int n = wave * 128 + j * 16 + (lane & 15);
      float bv = bias[n];
      int mb = (lane >> 4) << 3;
#pragma unroll
      for (int i = 0; i < 8; ++i)
        y1[(size_t)(tok0 + mt * 16 + mb + i) * DIM + n] =
            f2bf(silu(acc[mt][j][i] + bv));
    }
  }
}

// ---------------------------------------------------------------------------
// trans_mlp2: y2 = y1 @ tr_w2 + b2 ; out += LN(y2)*gamma+beta (out holds xn).
// M tile = 32 tokens; block owns full 1024-wide rows -> LN fused in LDS.
// LN tile aliases the A/B staging area (dead after the K loop's last barrier).
__global__ __launch_bounds__(256) void trans_mlp2_kernel(
    const u16* __restrict__ y1, const u16* __restrict__ w,
    const float* __restrict__ bias, const float* __restrict__ gamma,
    const float* __restrict__ beta, float* __restrict__ out) {
  extern __shared__ unsigned char smem[];
  u16*   As    = (u16*)smem;                          // 32*40 bf16
  u16*   Bt    = (u16*)(smem + 32 * A_STRIDE * 2);    // 1024*40 bf16
  float* ytile = (float*)smem;                        // 32*1028 f32 (alias)
  float* mrow  = ytile + 32 * YT_STRIDE;              // 32 f32
  float* rrow  = mrow + 32;                           // 32 f32
  const int tid = threadIdx.x, lane = tid & 31, wave = tid >> 5;
  const int tok0 = blockIdx.x * 32;
  v8f acc[2][8] = {};
  for (int k0 = 0; k0 < DIM; k0 += 32) {
    for (int e = tid; e < 32 * 32; e += 256) {
      int m = e >> 5, k = e & 31;
      As[m * A_STRIDE + k] = y1[(size_t)(tok0 + m) * DIM + k0 + k];
    }
    for (int q = tid; q < 32 * (DIM / 4); q += 256) {
      int k = q >> 8, n4 = (q & 255) << 2;
      us4 wv = *(const us4*)(w + (size_t)(k0 + k) * DIM + n4);
      Bt[(n4 + 0) * BT_STRIDE + k] = wv.x;
      Bt[(n4 + 1) * BT_STRIDE + k] = wv.y;
      Bt[(n4 + 2) * BT_STRIDE + k] = wv.z;
      Bt[(n4 + 3) * BT_STRIDE + k] = wv.w;
    }
    if (k0 + 32 < DIM)
      __builtin_prefetch(w + (size_t)(k0 + 32) * DIM + ((size_t)tid << 2), 0, 1);
    __syncthreads();
    v16bf a0 = load_a_frag(As, lane);
    v16bf a1 = load_a_frag(As + 16 * A_STRIDE, lane);
#pragma unroll
    for (int j = 0; j < 8; ++j) {
      v16bf b = load_b_frag(Bt, lane, wave * 128 + j * 16);
      acc[0][j] = wmma_bf16(a0, b, acc[0][j]);
      acc[1][j] = wmma_bf16(a1, b, acc[1][j]);
    }
    __syncthreads();          // As/Bt dead after the final iteration
  }
  // stage accumulators + bias into LDS tile (aliases As/Bt)
#pragma unroll
  for (int mt = 0; mt < 2; ++mt) {
#pragma unroll
    for (int j = 0; j < 8; ++j) {
      int n = wave * 128 + j * 16 + (lane & 15);
      float bv = bias[n];
      int mb = (lane >> 4) << 3;
#pragma unroll
      for (int i = 0; i < 8; ++i)
        ytile[(mt * 16 + mb + i) * YT_STRIDE + n] = acc[mt][j][i] + bv;
    }
  }
  __syncthreads();
  // per-row LN stats: 8 lanes per row (32 rows * 8 = 256 threads),
  // xor-reduce within aligned 8-groups of the wave
  {
    int r = tid >> 3, jj = tid & 7;
    float s = 0.f, sq = 0.f;
    for (int c = jj; c < DIM; c += 8) {
      float v = ytile[r * YT_STRIDE + c];
      s += v; sq += v * v;
    }
#pragma unroll
    for (int off = 4; off; off >>= 1) {
      s  += __shfl_xor(s, off);
      sq += __shfl_xor(sq, off);
    }
    if (jj == 0) {
      float m   = s * (1.0f / DIM);
      float var = sq * (1.0f / DIM) - m * m;
      mrow[r] = m;
      rrow[r] = rsqrtf(var + 1e-5f);
    }
  }
  __syncthreads();
  // out += LN(y)*gamma+beta  (coalesced)
  for (int e = tid; e < 32 * DIM; e += 256) {
    int r = e >> 10, c = e & 1023;
    float v = (ytile[r * YT_STRIDE + c] - mrow[r]) * rrow[r] * gamma[c] + beta[c];
    size_t idx = (size_t)(tok0 + r) * DIM + c;
    out[idx] = out[idx] + v;
  }
}

// ---------------------------------------------------------------------------
extern "C" void kernel_launch(void* const* d_in, const int* in_sizes, int n_in,
                              void* d_out, int out_size, void* d_ws, size_t ws_size,
                              hipStream_t stream) {
  (void)in_sizes; (void)n_in; (void)out_size; (void)ws_size;
  const float* x        = (const float*)d_in[0];
  const float* pred_w1  = (const float*)d_in[1];
  const float* pred_b1  = (const float*)d_in[2];
  const float* pred_w2  = (const float*)d_in[3];
  const float* pred_b2  = (const float*)d_in[4];
  const float* norm_g   = (const float*)d_in[5];
  const float* norm_b   = (const float*)d_in[6];
  const float* emb      = (const float*)d_in[7];
  const float* tr_w1    = (const float*)d_in[8];
  const float* tr_b1    = (const float*)d_in[9];
  const float* tr_w2    = (const float*)d_in[10];
  const float* tr_b2    = (const float*)d_in[11];
  const float* tr_gamma = (const float*)d_in[12];
  const float* tr_beta  = (const float*)d_in[13];

  char* ws = (char*)d_ws;
  size_t o = 0;
  int* sid  = (int*)(ws + o);  o += (size_t)NTOK * 4;                 // 128 KB
  u16* w1b  = (u16*)(ws + o);  o += (size_t)1024 * 512 * 2;           // 1 MB
  u16* tw1b = (u16*)(ws + o);  o += (size_t)2048 * 1024 * 2;          // 4 MB
  u16* tw2b = (u16*)(ws + o);  o += (size_t)1024 * 1024 * 2;          // 2 MB
  u16* tbuf = (u16*)(ws + o);  o += (size_t)NTOK * TDIM * 2;          // 128 MB
  u16* y1   = (u16*)(ws + o);                                          // 64 MB

  float* out    = (float*)d_out;
  float* states = out + (size_t)NTOK * DIM;

  cvt_bf16_kernel<<<512, 256, 0, stream>>>(pred_w1, w1b, 1024 * 512);
  cvt_bf16_kernel<<<1024, 256, 0, stream>>>(tr_w1, tw1b, 2048 * 1024);
  cvt_bf16_kernel<<<1024, 256, 0, stream>>>(tr_w2, tw2b, 1024 * 1024);

  const int pred_stage = 16 * A_STRIDE * 2 + HID * BT_STRIDE * 2;     // 42240
  const int pred_hbuf  = 16 * 520 * 4;                                // 33280
  const int pred_lds   = pred_stage > pred_hbuf ? pred_stage : pred_hbuf;
  pred_kernel<<<NTOK / 16, 256, pred_lds, stream>>>(x, w1b, pred_b1, pred_w2,
                                                    pred_b2, states, sid);

  ln_gather_kernel<<<NTOK / 8, 256, 0, stream>>>(x, sid, norm_g, norm_b, emb,
                                                 out, tbuf);

  const int mlp_stage = 32 * A_STRIDE * 2 + DIM * BT_STRIDE * 2;      // 84480
  trans_mlp1_kernel<<<NTOK / 32, 256, mlp_stage, stream>>>(tbuf, tw1b, tr_b1,
                                                           y1);

  const int ln_tile  = (32 * YT_STRIDE + 64) * 4;                     // 131840
  const int mlp2_lds = mlp_stage > ln_tile ? mlp_stage : ln_tile;
  trans_mlp2_kernel<<<NTOK / 32, 256, mlp2_lds, stream>>>(y1, tw2b, tr_b2,
                                                          tr_gamma, tr_beta,
                                                          out);
}
